// _LVCBlock_38345468018954
// MI455X (gfx1250) — compile-verified, matrix operand order checked by reading, not run on hardware
//
#include <hip/hip_runtime.h>
#include <hip/hip_bf16.h>
#include <math.h>

// ---------------------------------------------------------------------------
// LVC block for MI455X (gfx1250, wave32, WMMA)
//   CH=128, T=16384, B=8, 4 dilated layers (d = 1,2,4,8), k=3
// ---------------------------------------------------------------------------

#define CH        128
#define T_AUDIO   16384
#define BATCH     8
#define KSIZE     3
#define N_LAYERS  4
#define KC        (CH*CH*KSIZE*N_LAYERS)   // 196608
#define K_PER_L   (CH*CH*KSIZE)            // 49152
#define KTOT      (KC + CH*N_LAYERS)       // 197120 rows of w2

#define TCH       128        // t columns per block
#define MAXROWS   144        // TCH + 2*max_dil
#define LDSTR     136        // halfs per LDS row (272B, 16B aligned, padded)

typedef _Float16 v16hf __attribute__((ext_vector_type(16)));
typedef _Float16 v8hf  __attribute__((ext_vector_type(8)));
typedef float    v8ff  __attribute__((ext_vector_type(8)));

union V16U { v16hf v; v8hf h[2]; };

// ---------------------------------------------------------------------------
// Stage 1: cond net -> h_mean (8 x 64).  Tiny; one block per batch sample.
// ---------------------------------------------------------------------------
__global__ __launch_bounds__(256) void cond_net_kernel(
    const float* __restrict__ cond, const float* __restrict__ w0,
    const float* __restrict__ b0,   const float* __restrict__ w1,
    const float* __restrict__ b1,   float* __restrict__ hmean)
{
    __shared__ float sc[80*32];
    __shared__ float h1[64*32];
    __shared__ float h2[64*32];
    const int b = blockIdx.x, tid = threadIdx.x;

    for (int i = tid; i < 80*32; i += 256) sc[i] = cond[b*80*32 + i];
    __syncthreads();

    for (int i = tid; i < 64*32; i += 256) {           // conv k=5 pad=2 + gelu
        int o = i >> 5, t = i & 31;
        float s = b0[o];
        for (int c = 0; c < 80; ++c)
            #pragma unroll
            for (int k = 0; k < 5; ++k) {
                int ti = t + k - 2;
                if (ti >= 0 && ti < 32) s += w0[(o*80 + c)*5 + k] * sc[c*32 + ti];
            }
        h1[i] = 0.5f * s * (1.f + erff(s * 0.70710678118654752f));
    }
    __syncthreads();

    for (int i = tid; i < 64*32; i += 256) {           // conv k=3 pad=1 + gelu
        int o = i >> 5, t = i & 31;
        float s = b1[o];
        for (int c = 0; c < 64; ++c)
            #pragma unroll
            for (int k = 0; k < 3; ++k) {
                int ti = t + k - 1;
                if (ti >= 0 && ti < 32) s += w1[(o*64 + c)*3 + k] * h1[c*32 + ti];
            }
        h2[i] = 0.5f * s * (1.f + erff(s * 0.70710678118654752f));
    }
    __syncthreads();

    if (tid < 64) {                                    // mean over t (commutes with 1x1 conv)
        float s = 0.f;
        for (int t = 0; t < 32; ++t) s += h2[tid*32 + t];
        hmean[b*64 + tid] = s * (1.f/32.f);
    }
}

// ---------------------------------------------------------------------------
// Stage 2: weight generation GEMM (197120 x 64)·(64 x 8), scattered directly
// into the CDNA5 WMMA A-matrix lane layout (f16) + per-layer biases (f32).
// wblob[l][b][mtile][kt][lane][h]  (h = 16 halfs per lane = one v16hf A tile)
// ---------------------------------------------------------------------------
__global__ __launch_bounds__(256) void gen_weights_kernel(
    const float* __restrict__ w2, const float* __restrict__ b2,
    const float* __restrict__ hmean, _Float16* __restrict__ wblob,
    float* __restrict__ bias)
{
    __shared__ float hm[512];
    const int tid = threadIdx.x;
    for (int i = tid; i < 512; i += 256) hm[i] = hmean[i];
    __syncthreads();

    const int o = blockIdx.x * 256 + tid;
    if (o >= KTOT) return;

    const float* wrow = w2 + (size_t)o * 64;
    float dots[BATCH];
    #pragma unroll
    for (int b = 0; b < BATCH; ++b) dots[b] = b2[o];
    for (int c = 0; c < 64; ++c) {
        float w = wrow[c];
        #pragma unroll
        for (int b = 0; b < BATCH; ++b) dots[b] += w * hm[b*64 + c];
    }

    if (o < KC) {
        int l   = o / K_PER_L;
        int r   = o - l * K_PER_L;
        int co  = r / (CH*KSIZE);
        int rem = r - co * (CH*KSIZE);
        int ci  = rem / KSIZE;
        int tap = rem - ci * KSIZE;
        int k   = tap * CH + ci;                 // K dim = tap-major
        int kt  = k >> 5, kk = k & 31;
        int grp = (kk >> 3) & 1;                 // half-wave group
        int h   = (kk & 7) + ((kk >> 4) << 3);   // half index within lane
        int mtile = co >> 4, m = co & 15;
        int lane  = grp * 16 + m;
        #pragma unroll
        for (int b = 0; b < BATCH; ++b) {
            size_t idx = ((((size_t)(l*BATCH + b)*8 + mtile)*12 + kt)*32 + lane)*16 + h;
            wblob[idx] = (_Float16)dots[b];
        }
    } else {
        int idx = o - KC;
        int l = idx >> 7, co = idx & 127;
        #pragma unroll
        for (int b = 0; b < BATCH; ++b)
            bias[(l*BATCH + b)*CH + co] = dots[b];
    }
}

// ---------------------------------------------------------------------------
// Stage 3: dilated 128->128 k=3 conv as WMMA GEMM + bias + residual (+snake).
// Grid: (128 t-tiles, 8 batch), 256 threads = 8 waves.
// Wave tiling: 2 co-tiles x 4 t-tiles per wave (waves 0-3: t-tiles 0-3,
// waves 4-7: t-tiles 4-7).  Each B tile from LDS feeds TWO back-to-back
// WMMAs (A0*B, A1*B), halving LDS bandwidth per WMMA to match the matrix
// pipe rate.  96 v_wmma_f32_16x16x32_f16 per wave, fully unrolled.
// ---------------------------------------------------------------------------
__global__ __launch_bounds__(256) void lvc_conv_kernel(
    const float* __restrict__ xin, float* __restrict__ xout,
    const _Float16* __restrict__ wblob, const float* __restrict__ bias,
    const float* __restrict__ alpha, int dil, int snake)
{
    __shared__ alignas(16) _Float16 xT[MAXROWS * LDSTR];

    const int b     = blockIdx.y;
    const int tbase = blockIdx.x * TCH;
    const int tid   = threadIdx.x;
    const int rows  = TCH + 2*dil;
    const long xoff = (long)b * (CH * (long)T_AUDIO);

    // ---- stage x slab into LDS, transposed to [t][ci], f32 -> f16 ----------
    for (int idx = tid; idx < MAXROWS*CH; idx += 256) {
        int ci = idx / MAXROWS;               // constant divisor -> magic mul
        int tr = idx - ci * MAXROWS;
        if (tr < rows) {
            int t = tbase - dil + tr;
            float v = ((unsigned)t < (unsigned)T_AUDIO)
                        ? xin[xoff + (long)ci * T_AUDIO + t] : 0.f;
            xT[tr*LDSTR + ci] = (_Float16)v;
        }
    }
    __syncthreads();

    const int wave  = tid >> 5;
    const int lane  = tid & 31;
    const int n16   = lane & 15;      // N column within tile
    const int grp   = lane >> 4;      // half-wave group
    const int mpair = wave & 3;       // co-tile pair {2*mpair, 2*mpair+1}
    const int nhalf = wave >> 2;      // t-tile range base = nhalf*4

    v8ff acc[2][4];
    #pragma unroll
    for (int m = 0; m < 2; ++m)
        #pragma unroll
        for (int n = 0; n < 4; ++n)
            #pragma unroll
            for (int r = 0; r < 8; ++r) acc[m][n][r] = 0.f;

    const _Float16* wbase0 =
        wblob + ((size_t)(b*8 + 2*mpair) * 12) * (32*16) + (size_t)lane * 16;
    const _Float16* wbase1 = wbase0 + (size_t)12 * (32*16);

    // Per-tap LDS row base (bytes); includes this wave's t-tile half.
    // (kt&3)/n offsets become compile-time ds_load immediates after unroll.
    int rowbase[3];
    #pragma unroll
    for (int tap = 0; tap < 3; ++tap)
        rowbase[tap] = (n16 + nhalf*64 + tap*dil) * (LDSTR*2) + grp * 16;

    const char* ldsb = (const char*)xT;

    auto loadA = [&](const _Float16* wb, int kt, V16U& dst) {
        const _Float16* ap = wb + (size_t)kt * (32*16);
        dst.h[0] = *(const v8hf*)(ap);
        dst.h[1] = *(const v8hf*)(ap + 8);
    };
    auto loadB = [&](int kt, int n, V16U& dst) {
        const int tap = kt >> 2;
        const char* p = ldsb + rowbase[tap] + (kt & 3)*64 + n*(16*LDSTR*2);
        dst.h[0] = *(const v8hf*)(p);
        dst.h[1] = *(const v8hf*)(p + 32);
    };

    V16U a0[2], a1[2], bm[2];
    loadA(wbase0, 0, a0[0]);
    loadA(wbase1, 0, a1[0]);
    loadB(0, 0, bm[0]);

    #pragma unroll
    for (int kt = 0; kt < 12; ++kt) {
        if (kt < 11) {                                  // prefetch next A pair
            loadA(wbase0, kt + 1, a0[(kt + 1) & 1]);
            loadA(wbase1, kt + 1, a1[(kt + 1) & 1]);
        }
        #pragma unroll
        for (int n = 0; n < 4; ++n) {
            int nn = n + 1, nkt = kt;                   // prefetch next B
            if (nn == 4) { nn = 0; nkt = kt + 1; }
            if (nkt < 12) loadB(nkt, nn, bm[(n + 1) & 1]);
            // two WMMAs share one B tile -> 2x LDS bandwidth reuse
            acc[0][n] = __builtin_amdgcn_wmma_f32_16x16x32_f16(
                            false, a0[kt & 1].v, false, bm[n & 1].v,
                            (short)0, acc[0][n], false, false);
            acc[1][n] = __builtin_amdgcn_wmma_f32_16x16x32_f16(
                            false, a1[kt & 1].v, false, bm[n & 1].v,
                            (short)0, acc[1][n], false, false);
        }
    }

    // ---- epilogue: bias + residual (+ snake) ------------------------------
    const float* bl = bias + b * CH;
    #pragma unroll
    for (int m = 0; m < 2; ++m)
        #pragma unroll
        for (int n = 0; n < 4; ++n) {
            const int t = tbase + (nhalf*4 + n)*16 + n16;
            #pragma unroll
            for (int r = 0; r < 8; ++r) {
                const int co = (2*mpair + m)*16 + grp*8 + r;  // D: M = r (+8 grp1)
                const long o = xoff + (long)co * T_AUDIO + t;
                float v = acc[m][n][r] + bl[co] + xin[o];
                if (snake) {
                    float al = alpha[co];
                    float s  = sinf(al * v);
                    v = v + (1.0f / (al + 1e-8f)) * s * s;
                }
                xout[o] = v;
            }
        }
}

// ---------------------------------------------------------------------------
extern "C" void kernel_launch(void* const* d_in, const int* in_sizes, int n_in,
                              void* d_out, int out_size, void* d_ws, size_t ws_size,
                              hipStream_t stream)
{
    (void)in_sizes; (void)n_in; (void)out_size; (void)ws_size;

    const float* x     = (const float*)d_in[0];
    const float* cond  = (const float*)d_in[1];
    const float* w0    = (const float*)d_in[2];
    const float* b0    = (const float*)d_in[3];
    const float* w1    = (const float*)d_in[4];
    const float* b1    = (const float*)d_in[5];
    const float* w2    = (const float*)d_in[6];
    const float* b2    = (const float*)d_in[7];
    const float* alpha = (const float*)d_in[8];
    float* out = (float*)d_out;

    char* ws = (char*)d_ws;
    float*    hmean = (float*)ws;                       // 2 KB
    float*    bias  = (float*)(ws + 4096);              // 16 KB  (4 layers x 8 x 128)
    _Float16* wblob = (_Float16*)(ws + 4096 + 16384);   // 3 MB   (A-layout f16 kernels)
    float*    xbuf  = (float*)(ws + (size_t)(4u << 20));// 64 MB  ping buffer

    cond_net_kernel<<<BATCH, 256, 0, stream>>>(cond, w0, b0, w1, b1, hmean);
    gen_weights_kernel<<<(KTOT + 255)/256, 256, 0, stream>>>(w2, b2, hmean, wblob, bias);

    const size_t WL = (size_t)BATCH * 8 * 12 * 32 * 16;   // halfs per layer
    dim3 grid(T_AUDIO / TCH, BATCH);

    lvc_conv_kernel<<<grid, 256, 0, stream>>>(x,    xbuf, wblob + 0*WL, bias + 0*BATCH*CH, alpha, 1, 0);
    lvc_conv_kernel<<<grid, 256, 0, stream>>>(xbuf, out,  wblob + 1*WL, bias + 1*BATCH*CH, alpha, 2, 0);
    lvc_conv_kernel<<<grid, 256, 0, stream>>>(out,  xbuf, wblob + 2*WL, bias + 2*BATCH*CH, alpha, 4, 0);
    lvc_conv_kernel<<<grid, 256, 0, stream>>>(xbuf, out,  wblob + 3*WL, bias + 3*BATCH*CH, alpha, 8, 1);
}
